// Graph_attention_25572235280995
// MI455X (gfx1250) — compile-verified, matrix operand order checked by reading, not run on hardware
//
#include <hip/hip_runtime.h>
#include <hip/hip_bf16.h>

#define H 256
#define BB 32
#define TT 4096
#define EE 8192
#define D2 512   // 2H

typedef __attribute__((ext_vector_type(16))) __bf16        v16bf;
typedef __attribute__((ext_vector_type(8)))  float         v8f;
typedef __attribute__((ext_vector_type(8)))  unsigned int  v8u;

__device__ __forceinline__ unsigned short f2bf(float x) {
    unsigned int u = __float_as_uint(x);
    unsigned int r = u + 0x7FFFu + ((u >> 16) & 1u);   // round-to-nearest-even
    return (unsigned short)(r >> 16);
}
__device__ __forceinline__ unsigned int pk2(float lo, float hi) {
    return (unsigned int)f2bf(lo) | ((unsigned int)f2bf(hi) << 16);
}

// Hardware tanh: gfx1250 has V_TANH_F32 (TRANS pipe). Use the builtin when the
// toolchain declares it, else emit the instruction directly.
__device__ __forceinline__ float htanh(float x) {
#if __has_builtin(__builtin_amdgcn_tanhf)
    return __builtin_amdgcn_tanhf(x);
#elif __has_builtin(__builtin_amdgcn_tanh_f32)
    return __builtin_amdgcn_tanh_f32(x);
#else
    float y;
    asm volatile("v_tanh_f32 %0, %1\n\t"
                 "s_delay_alu instid0(TRANS32_DEP_1)"
                 : "=v"(y) : "v"(x));
    return y;
#endif
}

// ---------------------------------------------------------------------------
// Pack W2 (= W_attn[:, 256:512], f32) into bf16 in the exact v_wmma B-operand
// layout: per (ntile, kchunk) a 32-lane x 16-bf16 block.
// Lane L (0-15): col n = nt*16+L, K = kbase+{0..7, 16..23}
// Lane L+16   : same col,        K = kbase+8+{0..7, 16..23}   (kbase = kc*32)
// ---------------------------------------------------------------------------
__global__ void packb_kernel(const float* __restrict__ W, unsigned short* __restrict__ Bp) {
    int blk  = blockIdx.x;            // 0..255 : nt*8 + kc
    int nt   = blk >> 3;
    int kc   = blk & 7;
    int lane = threadIdx.x;           // 0..31
    int n     = nt * 16 + (lane & 15);
    int kbase = kc * 32 + ((lane >> 4) ? 8 : 0);
    const float* wrow = W + (size_t)n * D2 + H;   // right half of row n
    unsigned short* dst = Bp + ((size_t)blk * 32 + lane) * 16;
    #pragma unroll
    for (int j = 0; j < 8; ++j) dst[j]     = f2bf(wrow[kbase + j]);
    #pragma unroll
    for (int j = 0; j < 8; ++j) dst[8 + j] = f2bf(wrow[kbase + 16 + j]);
}

// P[b][n] = b_attn[n] + hidden[b,:] . W_attn[n, 0:256]
__global__ void p_kernel(const float* __restrict__ hidden, const float* __restrict__ W,
                         const float* __restrict__ bvec, float* __restrict__ P) {
    int b = blockIdx.x;       // 32
    int n = threadIdx.x;      // 512
    const float* h = hidden + b * H;
    const float* w = W + (size_t)n * D2;
    float acc = bvec[n];
    #pragma unroll 4
    for (int k = 0; k < H; ++k) acc = fmaf(h[k], w[k], acc);
    P[b * D2 + n] = acc;
}

// ---------------------------------------------------------------------------
// Main fused kernel. Per wave: rows = (b, t0..t0+31) as TWO 16-row M-tiles
// sharing every B chunk (halves L2 traffic, 2 wmma per B load), GEMM over
// K=256 with v_wmma_f32_16x16x32_bf16, fused  +P, tanh, dot-with-v  epilogue.
// ---------------------------------------------------------------------------
__global__ void __launch_bounds__(256)
scores_kernel(const float* __restrict__ enc, const v8u* __restrict__ Bq,
              const float* __restrict__ P, const float* __restrict__ v,
              float* __restrict__ scores) {
    int wave = blockIdx.x * 8 + (threadIdx.x >> 5);   // 0..4095
    int lane = threadIdx.x & 31;
    int b    = wave >> 7;                 // 0..31
    int t0   = (wave & 127) << 5;         // 0..4064 step 32
    int mrow = lane & 15;
    int kgrp = lane >> 4;

    // --- Load & bf16-pack both A tiles: 2 x (16 rows x 256 K), ISA A layout --
    const float* rowp0 = enc + ((size_t)(t0 + mrow)      * BB + b) * H + kgrp * 8;
    const float* rowp1 = enc + ((size_t)(t0 + 16 + mrow) * BB + b) * H + kgrp * 8;
    v16bf a0[8], a1[8];
    #pragma unroll
    for (int kc = 0; kc < 8; ++kc) {
        const float* p0 = rowp0 + kc * 32;
        const float* p1 = rowp1 + kc * 32;
        v8u u0, u1;
        #pragma unroll
        for (int j = 0; j < 4; ++j) { u0[j] = pk2(p0[2 * j],     p0[2 * j + 1]);
                                      u1[j] = pk2(p1[2 * j],     p1[2 * j + 1]); }
        #pragma unroll
        for (int j = 4; j < 8; ++j) { u0[j] = pk2(p0[8 + 2 * j], p0[8 + 2 * j + 1]);
                                      u1[j] = pk2(p1[8 + 2 * j], p1[8 + 2 * j + 1]); }
        a0[kc] = __builtin_bit_cast(v16bf, u0);
        a1[kc] = __builtin_bit_cast(v16bf, u1);
    }

    float sacc0[8], sacc1[8];
    #pragma unroll
    for (int i = 0; i < 8; ++i) { sacc0[i] = 0.f; sacc1[i] = 0.f; }

    const float* Pb = P + b * D2;
    for (int nt = 0; nt < 32; ++nt) {
        const v8u* bp = Bq + ((size_t)nt * 8 * 32 + lane);
        if (nt < 31) __builtin_prefetch(bp + 8 * 32, 0, 1);   // global_prefetch next N-tile
        v8f c0 = {}, c1 = {};
        #pragma unroll
        for (int kc = 0; kc < 8; ++kc) {
            v16bf bv = __builtin_bit_cast(v16bf, bp[kc * 32]);
            c0 = __builtin_amdgcn_wmma_f32_16x16x32_bf16(
                     false, a0[kc], false, bv, (short)0, c0, false, false);
            c1 = __builtin_amdgcn_wmma_f32_16x16x32_bf16(
                     false, a1[kc], false, bv, (short)0, c1, false, false);
        }
        // epilogue for this lane's column n (TRANS-pipe tanh, co-executes)
        int   n  = nt * 16 + (lane & 15);
        float pn = Pb[n];
        float vn = v[n];
        #pragma unroll
        for (int i = 0; i < 8; ++i) {
            sacc0[i] += vn * htanh(c0[i] + pn);
            sacc1[i] += vn * htanh(c1[i] + pn);
        }
    }

    // reduce the 16 column-lanes within each half of the wave
    #pragma unroll
    for (int i = 0; i < 8; ++i) {
        float s0 = sacc0[i], s1 = sacc1[i];
        #pragma unroll
        for (int off = 1; off < 16; off <<= 1) {
            s0 += __shfl_xor(s0, off, 32);
            s1 += __shfl_xor(s1, off, 32);
        }
        sacc0[i] = s0; sacc1[i] = s1;
    }
    if ((lane & 15) == 0) {
        #pragma unroll
        for (int i = 0; i < 8; ++i) {
            scores[b * TT + t0 +      kgrp * 8 + i] = sacc0[i];
            scores[b * TT + t0 + 16 + kgrp * 8 + i] = sacc1[i];
        }
    }
}

// In-place softmax over T per batch row.
__global__ void softmax_kernel(float* __restrict__ w) {
    __shared__ float red[256];
    int b = blockIdx.x, tid = threadIdx.x;
    float* row = w + b * TT;
    float m = -1e30f;
    for (int t = tid; t < TT; t += 256) m = fmaxf(m, row[t]);
    red[tid] = m; __syncthreads();
    for (int s = 128; s > 0; s >>= 1) { if (tid < s) red[tid] = fmaxf(red[tid], red[tid + s]); __syncthreads(); }
    m = red[0]; __syncthreads();
    float sum = 0.f;
    for (int t = tid; t < TT; t += 256) sum += __expf(row[t] - m);
    red[tid] = sum; __syncthreads();
    for (int s = 128; s > 0; s >>= 1) { if (tid < s) red[tid] += red[tid + s]; __syncthreads(); }
    float inv = 1.0f / red[0];
    for (int t = tid; t < TT; t += 256) row[t] = __expf(row[t] - m) * inv;
}

__global__ void zero_kernel(float* __restrict__ out, int n) {
    int i = blockIdx.x * blockDim.x + threadIdx.x;
    if (i < n) out[i] = 0.f;
}

// out[b, dst] += 0.1 * w[b, src]  for edges with dst != src+1
__global__ void scatter_kernel(const int* __restrict__ src, const int* __restrict__ dst,
                               const float* __restrict__ w, float* __restrict__ out) {
    int i = blockIdx.x * blockDim.x + threadIdx.x;
    if (i >= BB * EE) return;
    int b = i >> 13;                    // E = 8192
    int s = src[i], d = dst[i];
    if (d != s + 1)
        atomicAdd(&out[b * TT + d], w[b * TT + s] * 0.1f);
}

extern "C" void kernel_launch(void* const* d_in, const int* in_sizes, int n_in,
                              void* d_out, int out_size, void* d_ws, size_t ws_size,
                              hipStream_t stream) {
    const float* hidden = (const float*)d_in[0];
    const float* enc    = (const float*)d_in[1];
    const int*   esrc   = (const int*)d_in[2];
    const int*   edst   = (const int*)d_in[3];
    const float* W      = (const float*)d_in[4];
    const float* bvec   = (const float*)d_in[5];
    const float* v      = (const float*)d_in[6];
    float* out = (float*)d_out;

    char* ws = (char*)d_ws;
    unsigned short* Bpack = (unsigned short*)ws;            // 256 KB : bf16 B operand tiles
    float* P    = (float*)(ws + 262144);                    //  64 KB : hidden-side pre-acts
    float* wbuf = (float*)(ws + 262144 + 65536);            // 512 KB : scores -> softmax w

    packb_kernel  <<<256, 32, 0, stream>>>(W, Bpack);
    p_kernel      <<<BB, D2, 0, stream>>>(hidden, W, bvec, P);
    scores_kernel <<<512, 256, 0, stream>>>(enc, (const v8u*)Bpack, P, v, wbuf);
    softmax_kernel<<<BB, 256, 0, stream>>>(wbuf);
    zero_kernel   <<<(BB * TT + 255) / 256, 256, 0, stream>>>(out, BB * TT);
    scatter_kernel<<<(BB * EE + 255) / 256, 256, 0, stream>>>(esrc, edst, wbuf, out);
}
